// CVXPolicy_MultiQuadcopter_14216341749945
// MI455X (gfx1250) — matrix operand order repaired and checked by reading, not compiled, not run on hardware
//
#include <hip/hip_runtime.h>
#include <hip/hip_bf16.h>

typedef __attribute__((ext_vector_type(16))) _Float16 v16h;
typedef __attribute__((ext_vector_type(8)))  float    v8f;
typedef __attribute__((ext_vector_type(4)))  float    v4f;

#define NUM_AGENTS_ 256
#define HIDDEN_ 100
#define BATCH_ 8192
#define STATE_DIM_ 3072
#define CONTROL_DIM_ 1024

static __device__ __forceinline__ v8f wmma_f16(v16h a, v16h b, v8f c) {
  // (neg_a, A, neg_b, B, c_mod, C, reuse_a, reuse_b)
  return __builtin_amdgcn_wmma_f32_16x16x32_f16(false, a, false, b, (short)0, c, false, false);
}

// ---------------------------------------------------------------------------
// Prep: split f32 weight matrix into f16 hi/lo planes, pre-swizzled into the
// 16x16x32 WMMA B-fragment order used here (contiguous k, matching the A side):
//   fragment (kc, nt): lane L holds col = nt*16 + (L&15),
//                      K = kc*32 + (L>>4)*16 + e  for e = 0..15 (v16h).
// dst layout: [kc][nt][plane(hi=0,lo=1)][lane][e]  (_Float16)
// ---------------------------------------------------------------------------
__global__ void prep_bfrag(const float* __restrict__ src, int Ksrc, int rowOff,
                           int ldsrc, int Nsrc, int kChunks, int NTtot,
                           _Float16* __restrict__ dst) {
  int idx = blockIdx.x * blockDim.x + threadIdx.x;
  int total = kChunks * NTtot * 512;
  if (idx >= total) return;
  int e    = idx & 15;
  int lane = (idx >> 4) & 31;
  int t2   = idx >> 9;
  int nt   = t2 % NTtot;
  int kc   = t2 / NTtot;
  int K    = kc * 32 + ((lane >> 4) << 4) + e;
  int col  = nt * 16 + (lane & 15);
  float v = 0.0f;
  if (K < Ksrc && col < Nsrc) v = src[(long)(K + rowOff) * ldsrc + col];
  _Float16 hi = (_Float16)v;
  _Float16 lo = (_Float16)(v - (float)hi);
  long base = ((long)kc * NTtot + nt) * 1024 + lane * 16 + e;
  dst[base]       = hi;   // hi plane
  dst[base + 512] = lo;   // lo plane
}

// Reduced GEMM3 weights: Wc (100 x 1024) folded from W3 (100 x 3072):
//   c0 = (p6+p7+p8)/MASS, c1 = p9, c2 = p10, c3 = p11   (per agent)
__global__ void prep_wc(const float* __restrict__ W3, _Float16* __restrict__ dst) {
  int idx = blockIdx.x * blockDim.x + threadIdx.x;   // kChunks=4, NTtot=64
  if (idx >= 4 * 64 * 512) return;
  int e    = idx & 15;
  int lane = (idx >> 4) & 31;
  int t2   = idx >> 9;
  int nt   = t2 % 64;
  int kc   = t2 / 64;
  int K    = kc * 32 + ((lane >> 4) << 4) + e;
  int col  = nt * 16 + (lane & 15);
  float v = 0.0f;
  if (K < HIDDEN_) {
    int a = col >> 2, j = col & 3;
    const float* r = W3 + (long)K * STATE_DIM_ + a * 12;
    v = (j == 0) ? (r[6] + r[7] + r[8]) * 2.0f /* 1/MASS */ : r[8 + j];
  }
  _Float16 hi = (_Float16)v;
  _Float16 lo = (_Float16)(v - (float)hi);
  long base = ((long)kc * 64 + nt) * 1024 + lane * 16 + e;
  dst[base]       = hi;
  dst[base + 512] = lo;
}

__global__ void prep_bc(const float* __restrict__ b3, float* __restrict__ bc) {
  int col = blockIdx.x * blockDim.x + threadIdx.x;
  if (col >= CONTROL_DIM_) return;
  int a = col >> 2, j = col & 3;
  const float* r = b3 + a * 12;
  bc[col] = (j == 0) ? (r[6] + r[7] + r[8]) * 2.0f : r[8 + j];
}

// ---------------------------------------------------------------------------
// Split-f16 WMMA GEMM: one wave computes a 16 x (NT*16) strip of C.
//   A: f32 row-major (M x lda), converted on the fly to f16 hi+lo fragments.
//   Bfrag: pre-swizzled f16 hi/lo fragments (L2-resident).
//   acc += Ahi*Bhi + Ahi*Blo + Alo*Bhi  (f32 accumulate) -> ~22-bit mantissa.
// mode: 0 = tanh(acc + bias)    (layer 2)
//       1 = tanh(acc + bias + t[row]*w0row[col])   (layer 1, folds t column)
//       2 = acc + bias          (layer 3 -> c)
// padCols: zero-fill C columns [NT*16, padCols) so K-padding stays exact.
// ---------------------------------------------------------------------------
template <int NT>
__global__ void wgemm_split(const float* __restrict__ A, int lda, int kChunks,
                            const _Float16* __restrict__ Bfrag, int NTtot, int ntGroups,
                            const float* __restrict__ bias, int biasN,
                            const float* __restrict__ w0row, const float* __restrict__ tvec,
                            float* __restrict__ C, int ldc, int mode, int padCols) {
  const int lane   = threadIdx.x & 31;
  const int wid    = blockIdx.x * (blockDim.x >> 5) + (threadIdx.x >> 5);
  const int mtile  = wid / ntGroups;
  const int ntbase = (wid - mtile * ntGroups) * NT;
  const int row    = mtile * 16 + (lane & 15);
  const int khalf  = (lane >> 4) * 16;         // A-fragment K sub-offset
  const float* arow = A + (long)row * lda + khalf;

  v8f acc[NT];
#pragma unroll
  for (int i = 0; i < NT; ++i) acc[i] = (v8f){};

  for (int kc = 0; kc < kChunks; ++kc) {
    const float* ap = arow + kc * 32;
    v4f a0 = *(const v4f*)(ap + 0);
    v4f a1 = *(const v4f*)(ap + 4);
    v4f a2 = *(const v4f*)(ap + 8);
    v4f a3 = *(const v4f*)(ap + 12);
    v16h ahi, alo;
#pragma unroll
    for (int e = 0; e < 4; ++e) {
      float x0 = a0[e], x1 = a1[e], x2 = a2[e], x3 = a3[e];
      _Float16 h0 = (_Float16)x0, h1 = (_Float16)x1;
      _Float16 h2 = (_Float16)x2, h3 = (_Float16)x3;
      ahi[e] = h0; ahi[4 + e] = h1; ahi[8 + e] = h2; ahi[12 + e] = h3;
      alo[e]      = (_Float16)(x0 - (float)h0);
      alo[4 + e]  = (_Float16)(x1 - (float)h1);
      alo[8 + e]  = (_Float16)(x2 - (float)h2);
      alo[12 + e] = (_Float16)(x3 - (float)h3);
    }
    const _Float16* bbase = Bfrag + ((long)kc * NTtot + ntbase) * 1024 + lane * 16;
#pragma unroll
    for (int nt = 0; nt < NT; ++nt) {
      const _Float16* bp = bbase + (long)nt * 1024;
      v16h bhi = *(const v16h*)bp;
      v16h blo = *(const v16h*)(bp + 512);
      acc[nt] = wmma_f16(ahi, bhi, acc[nt]);
      acc[nt] = wmma_f16(ahi, blo, acc[nt]);
      acc[nt] = wmma_f16(alo, bhi, acc[nt]);
    }
  }

  // Epilogue. C/D layout: lane L, acc element v -> row rbase+v, col (L&15).
  const int rbase = mtile * 16 + (lane >> 4) * 8;
  float tv[8];
  if (mode == 1) {
#pragma unroll
    for (int v = 0; v < 8; ++v) tv[v] = tvec[rbase + v];
  }
#pragma unroll
  for (int nt = 0; nt < NT; ++nt) {
    int col = (ntbase + nt) * 16 + (lane & 15);
    float bb = (col < biasN) ? bias[col] : 0.0f;
    float w0 = 0.0f;
    if (mode == 1 && col < biasN) w0 = w0row[col];
#pragma unroll
    for (int v = 0; v < 8; ++v) {
      float x = acc[nt][v] + bb;
      if (mode == 1) x += tv[v] * w0;
      if (mode != 2) x = tanhf(x);
      C[(long)(rbase + v) * ldc + col] = x;
    }
  }
  if (padCols > NT * 16) {
    for (int col = NT * 16 + (lane & 15); col < padCols; col += 16) {
#pragma unroll
      for (int v = 0; v < 8; ++v) C[(long)(rbase + v) * ldc + col] = 0.0f;
    }
  }
}

// ---------------------------------------------------------------------------
// Per-row: s = ||c||^2, w = LambertW(s*N) via Halley, k = sqrt(w*N/s),
// u = -k*c (in place). One block per row.
// ---------------------------------------------------------------------------
__global__ void finalize_lambert(float* __restrict__ out) {
  __shared__ float red[256];
  __shared__ float kk;
  const int row = blockIdx.x;
  float* c = out + (long)row * CONTROL_DIM_;
  float s = 0.0f;
  for (int i = threadIdx.x; i < CONTROL_DIM_; i += 256) {
    float v = c[i];
    s += v * v;
  }
  red[threadIdx.x] = s;
  __syncthreads();
  for (int st = 128; st > 0; st >>= 1) {
    if (threadIdx.x < st) red[threadIdx.x] += red[threadIdx.x + st];
    __syncthreads();
  }
  if (threadIdx.x == 0) {
    float S = red[0];
    float x = S * (float)NUM_AGENTS_;   // s/(2*alpha) = s*N
    float w = log1pf(x);
    for (int it = 0; it < 30; ++it) {
      float ew = expf(w);
      float f = w * ew - x;
      float denom = ew * (w + 1.0f) - (w + 2.0f) * f / (2.0f * (w + 1.0f));
      w = w - f / denom;
    }
    kk = (S > 0.0f) ? sqrtf(w * (float)NUM_AGENTS_ / S) : 0.0f;
  }
  __syncthreads();
  float k = kk;
  for (int i = threadIdx.x; i < CONTROL_DIM_; i += 256) c[i] = -k * c[i];
}

// ---------------------------------------------------------------------------
extern "C" void kernel_launch(void* const* d_in, const int* in_sizes, int n_in,
                              void* d_out, int out_size, void* d_ws, size_t ws_size,
                              hipStream_t stream) {
  const float* z  = (const float*)d_in[0];   // (8192, 3072)
  const float* t  = (const float*)d_in[1];   // (8192, 1)
  const float* W1 = (const float*)d_in[2];   // (3073, 100)
  const float* b1 = (const float*)d_in[3];   // (100,)
  const float* W2 = (const float*)d_in[4];   // (100, 100)
  const float* b2 = (const float*)d_in[5];   // (100,)
  const float* W3 = (const float*)d_in[6];   // (100, 3072)
  const float* b3 = (const float*)d_in[7];   // (3072,)
  float* out = (float*)d_out;                // (8192, 1024)

  char* ws = (char*)d_ws;
  _Float16* B1f = (_Float16*)(ws);                       // 96*7*2*512 f16 = 1,376,256 B
  _Float16* B2f = (_Float16*)(ws + 1376256);             // 4*7*2*512  f16 =    57,344 B
  _Float16* B3f = (_Float16*)(ws + 1376256 + 57344);     // 4*64*2*512 f16 =   524,288 B
  float*    bc  = (float*)(ws + 1957888);                //  1024 f32       =     4,096 B
  float*    h1  = (float*)(ws + 1961984);                //  8192*128 f32   = 4,194,304 B
  float*    h2  = (float*)(ws + 6156288);                //  8192*128 f32   = 4,194,304 B
  // total workspace: 10,350,592 B

  // Weight prep (cheap; re-run every call for determinism).
  prep_bfrag<<<(96 * 7 * 512) / 256, 256, 0, stream>>>(W1, 3072, /*rowOff=*/1, 100, 100, 96, 7, B1f);
  prep_bfrag<<<(4 * 7 * 512) / 256, 256, 0, stream>>>(W2, 100, 0, 100, 100, 4, 7, B2f);
  prep_wc<<<(4 * 64 * 512) / 256, 256, 0, stream>>>(W3, B3f);
  prep_bc<<<4, 256, 0, stream>>>(b3, bc);

  // Layer 1: h1 = tanh([t,z] @ W1 + b1); t column folded into epilogue.
  wgemm_split<7><<<64, 256, 0, stream>>>(z, 3072, 96, B1f, 7, 1,
                                         b1, 100, W1 /*row 0*/, t,
                                         h1, 128, /*mode=*/1, /*padCols=*/128);
  // Layer 2: h2 = tanh(h1 @ W2 + b2)
  wgemm_split<7><<<64, 256, 0, stream>>>(h1, 128, 4, B2f, 7, 1,
                                         b2, 100, nullptr, nullptr,
                                         h2, 128, /*mode=*/0, /*padCols=*/128);
  // Layer 3 (reduced): c = h2 @ Wc + bc  -> d_out
  wgemm_split<4><<<1024, 256, 0, stream>>>(h2, 128, 4, B3f, 64, 16,
                                           bc, 1024, nullptr, nullptr,
                                           out, 1024, /*mode=*/2, /*padCols=*/0);
  // u* = -k(row) * c  (LambertW closed form), in place.
  finalize_lambert<<<8192, 256, 0, stream>>>(out);
}
